// DAAGCN_32186484916470
// MI455X (gfx1250) — matrix operand order repaired and checked by reading, not compiled, not run on hardware
//
#include <hip/hip_runtime.h>
#include <math.h>

// ---------------- problem constants ----------------
namespace {
constexpr int NN = 1024;   // nodes
constexpr int BB = 64;     // batch
constexpr int HH = 64;     // hidden
constexpr int CC = 65;     // DIN + H
constexpr int BC = BB * CC;      // 4160
constexpr int KC = 195;          // K * C = 3*65
constexpr int KP = 224;          // KC padded to mult of 32
constexpr int EE = 16;
constexpr int TT = 12;
}

// ---------------- WMMA plumbing ----------------
typedef __attribute__((ext_vector_type(16))) __bf16 v16bf;
typedef __attribute__((ext_vector_type(8)))  float  v8f;
typedef __attribute__((ext_vector_type(4)))  unsigned v4u;
typedef __attribute__((ext_vector_type(8)))  int      v8i;
typedef __attribute__((ext_vector_type(4)))  int      v4i;

#define DEV __device__ __forceinline__

#if defined(__has_builtin)
#  if __has_builtin(__builtin_amdgcn_tensor_load_to_lds)
#    define USE_TDM 1
#  endif
#endif
#ifndef USE_TDM
#  define USE_TDM 0
#endif

DEV unsigned short f2bf(float f) {
  unsigned u = __builtin_bit_cast(unsigned, f);
  u += 0x7FFFu + ((u >> 16) & 1u);   // round-to-nearest-even
  return (unsigned short)(u >> 16);
}
DEV float bf2f(unsigned short b) {
  unsigned u = ((unsigned)b) << 16;
  return __builtin_bit_cast(float, u);
}

// 16-bit A (16x32): lanes 0-15 hold M=lane, VGPR v -> K {2v,2v+1} for v<4,
// {2v+8,...} for v>=4; lanes 16-31 same M with K offset +8.
DEV int aKmap(int v, int lane) {
  return ((v < 4) ? 2 * v : 2 * v + 8) + ((lane >= 16) ? 8 : 0);
}
// 16-bit B (32x16): lanes 0-15 N=lane, same K mapping as A (K offset +8 for lanes 16-31).
DEV int bKmap(int v, int lane) {
  return ((v < 4) ? 2 * v : 2 * v + 8) + ((lane >= 16) ? 8 : 0);
}

struct Frag { union { unsigned u[8]; v16bf v; }; };

DEV v8f wmma_bf16(const Frag& a, const Frag& b, v8f c) {
  return __builtin_amdgcn_wmma_f32_16x16x32_bf16(false, a.v, false, b.v,
                                                 (short)0, c, false, false);
}

#if USE_TDM
// TDM 2D tile load: rows x 32 bf16 elements, row stride ld (elements), into LDS
// with pad: after every 64 B (one 32-elem row) insert 16 B -> LDS pitch 40 halves.
// D# per CDNA5 ISA ch.8: group0 {count=1, lds_addr, global_addr, type=2},
// group1 {data_size=1(2B), pad_enable, pad_interval=3(16 dw), pad_amount=3(4 dw),
//         tensor_dim0=32, tensor_dim1=rows, tile_dim0=32, tile_dim1=rows,
//         tensor_dim0_stride=ld}; groups 2/3 zero (2D tensor).
DEV void tdm_load_2d(unsigned lds_addr, const unsigned short* gptr, int rows, int ld) {
  unsigned long long ga = (unsigned long long)(const void*)gptr;
  v4u g0;
  g0[0] = 1u;                                        // count=1, user descriptor
  g0[1] = lds_addr;                                  // LDS byte address
  g0[2] = (unsigned)(ga & 0xFFFFFFFFu);              // global_addr[31:0]
  g0[3] = (unsigned)((ga >> 32) & 0x01FFFFFFu) | (2u << 30);  // addr[56:32] | type=2
  v8i g1;
  g1[0] = (int)((1u << 16) | (1u << 20) | (3u << 22) | (3u << 25)); // dsize|pad cfg
  g1[1] = (int)(32u << 16);                          // tensor_dim0 = 32
  g1[2] = (int)(((unsigned)rows & 0xFFFFu) << 16);   // tensor_dim1 = rows
  g1[3] = (int)(32u << 16);                          // tile_dim0 = 32
  g1[4] = rows;                                      // tile_dim1 = rows (tile_dim2=0)
  g1[5] = ld;                                        // tensor_dim0_stride
  g1[6] = 0;
  g1[7] = 0;
  v4i gz4 = {0, 0, 0, 0};
#if __clang_major__ >= 23
  v8i gz8 = {0, 0, 0, 0, 0, 0, 0, 0};
  __builtin_amdgcn_tensor_load_to_lds(g0, g1, gz4, gz4, gz8, 0);
#else
  __builtin_amdgcn_tensor_load_to_lds(g0, g1, gz4, gz4, 0);
#endif
}
#endif

// ---------------- utility kernels ----------------
__global__ void zero_f(float* p, int n) {
  int i = blockIdx.x * blockDim.x + threadIdx.x;
  if (i < n) p[i] = 0.f;
}

// e = LN(node_emb + te) for gate and upd params; fp32 copy + bf16 K=32 padded copy
__global__ void e_kernel(const float* __restrict__ node_emb,
                         const float* __restrict__ te,
                         const float* __restrict__ glng, const float* __restrict__ glnb,
                         const float* __restrict__ ulng, const float* __restrict__ ulnb,
                         float* __restrict__ eg, float* __restrict__ eu,
                         unsigned short* __restrict__ egp, unsigned short* __restrict__ eup) {
  int n = blockIdx.x * blockDim.x + threadIdx.x;
  if (n >= NN) return;
  float x[EE];
  float m = 0.f;
  for (int d = 0; d < EE; ++d) { x[d] = node_emb[n * EE + d] + te[d]; m += x[d]; }
  m *= (1.f / EE);
  float v = 0.f;
  for (int d = 0; d < EE; ++d) { float dd = x[d] - m; v += dd * dd; }
  v *= (1.f / EE);
  float inv = rsqrtf(v + 1e-12f);
  for (int d = 0; d < EE; ++d) {
    float xn = (x[d] - m) * inv;
    float g = xn * glng[d] + glnb[d];
    float u = xn * ulng[d] + ulnb[d];
    eg[n * EE + d] = g;  egp[n * 32 + d] = f2bf(g);
    eu[n * EE + d] = u;  eup[n * 32 + d] = f2bf(u);
  }
  for (int d = EE; d < 32; ++d) { egp[n * 32 + d] = 0; eup[n * 32 + d] = 0; }
}

// logits = e e^T via WMMA. ep: [1024][32] bf16 (zero padded K). one 16x16 tile per wave.
__global__ __launch_bounds__(256) void logits_kernel(const unsigned short* __restrict__ ep,
                                                     float* __restrict__ logits) {
  int wave = threadIdx.x >> 5, lane = threadIdx.x & 31;
  int id = blockIdx.x * 8 + wave;       // 4096 tiles total
  int rt = id >> 6, ct = id & 63;
  Frag a, b;
  int rrow = rt * 16 + (lane & 15);
  int crow = ct * 16 + (lane & 15);
  for (int v = 0; v < 8; ++v)
    a.u[v] = *reinterpret_cast<const unsigned*>(&ep[rrow * 32 + aKmap(v, lane)]);
  for (int v = 0; v < 8; ++v)
    b.u[v] = *reinterpret_cast<const unsigned*>(&ep[crow * 32 + bKmap(v, lane)]);
  v8f c = {};
  c = wmma_bf16(a, b, c);
  int mhi = (lane >= 16) ? 8 : 0;
  for (int r = 0; r < 8; ++r) {
    int gm = rt * 16 + r + mhi;
    int gn = ct * 16 + (lane & 15);
    logits[(size_t)gm * NN + gn] = c[r];
  }
}

// row softmax -> bf16 sup
__global__ __launch_bounds__(256) void softmax_kernel(const float* __restrict__ logits,
                                                      unsigned short* __restrict__ sup) {
  __shared__ float red[256];
  int row = blockIdx.x, tid = threadIdx.x;
  float vals[4];
  float vmax = -3.4e38f;
  for (int i = 0; i < 4; ++i) {
    vals[i] = logits[(size_t)row * NN + tid + i * 256];
    vmax = fmaxf(vmax, vals[i]);
  }
  red[tid] = vmax; __syncthreads();
  for (int s = 128; s > 0; s >>= 1) {
    if (tid < s) red[tid] = fmaxf(red[tid], red[tid + s]);
    __syncthreads();
  }
  vmax = red[0]; __syncthreads();
  float sum = 0.f;
  for (int i = 0; i < 4; ++i) { vals[i] = __expf(vals[i] - vmax); sum += vals[i]; }
  red[tid] = sum; __syncthreads();
  for (int s = 128; s > 0; s >>= 1) {
    if (tid < s) red[tid] += red[tid + s];
    __syncthreads();
  }
  float inv = 1.f / red[0];
  for (int i = 0; i < 4; ++i)
    sup[(size_t)row * NN + tid + i * 256] = f2bf(vals[i] * inv);
}

// pack xc = concat([x_t, (z*)state]) into xcT[bc][n] and xg[n][b][k=0 slot]
__global__ void pack_kernel(const float* __restrict__ source, int t,
                            const float* __restrict__ h, const float* __restrict__ z,
                            int useZ,
                            unsigned short* __restrict__ xcT,
                            unsigned short* __restrict__ xg) {
  size_t idx = (size_t)blockIdx.x * blockDim.x + threadIdx.x;  // over BC*NN
  int n = (int)(idx & (NN - 1));
  int bc = (int)(idx >> 10);
  int b = bc / CC, c = bc % CC;
  float val;
  if (c == 0) {
    val = source[((size_t)(b * TT + t)) * NN + n];
  } else {
    size_t hi = ((size_t)b * NN + n) * HH + (c - 1);
    float hv = h[hi];
    val = useZ ? z[hi] * hv : hv;
  }
  unsigned short bv = f2bf(val);
  xcT[(size_t)bc * NN + n] = bv;
  xg[((size_t)n * BB + b) * KP + c] = bv;  // k=0 slot: Kidx = c
}

// NT GEMM: D[m][ncol] = sum_k A[m][k] * Bt[ncol][k], bf16 in / f32 acc, WMMA.
// Block tile 128(M) x 64(N), 8 waves (4 M x 2 N), wave tile 32x32, K step 32.
// A/B strips staged in double-buffered LDS via TDM (tensor_load_to_lds) with
// hardware padding giving a conflict-free 40-half pitch; wave 0 issues the DMA
// and synchronizes with s_wait_tensorcnt.
// mode 0: write xg[m][b][kslot + c] and outT[ncol][m]    (xg1 = sup @ X)
// mode 1: val = 2*acc - xcT[ncol][m]; write xg[...]       (Chebyshev xg2)
__global__ __launch_bounds__(256) void gemm_nt(const unsigned short* __restrict__ A, int lda,
                                               const unsigned short* __restrict__ Bt, int ldb,
                                               int Kdim, int mode,
                                               const unsigned short* __restrict__ xcT,
                                               unsigned short* __restrict__ xg,
                                               unsigned short* __restrict__ outT,
                                               int kslot) {
  __shared__ unsigned short lsA[2][128 * 40];   // pitch 40 halves
  __shared__ unsigned short lsB[2][64 * 40];
  int tid = threadIdx.x;
  int wave = tid >> 5, lane = tid & 31;
  int wm = wave & 3, wn = wave >> 2;
  int blockM = blockIdx.y * 128;
  int blockN = blockIdx.x * 64;
  int steps = Kdim >> 5;

  v8f acc[2][2];
  for (int i = 0; i < 2; ++i)
    for (int j = 0; j < 2; ++j) acc[i][j] = v8f{};

#if USE_TDM
  if (wave == 0) {
    tdm_load_2d((unsigned)(size_t)&lsA[0][0], A + (size_t)blockM * lda, 128, lda);
    tdm_load_2d((unsigned)(size_t)&lsB[0][0], Bt + (size_t)blockN * ldb, 64, ldb);
  }
#endif

  for (int i = 0; i < steps; ++i) {
    int buf = i & 1;
#if USE_TDM
    if (wave == 0) {
      if (i + 1 < steps) {
        int k0n = (i + 1) << 5;
        tdm_load_2d((unsigned)(size_t)&lsA[buf ^ 1][0],
                    A + (size_t)blockM * lda + k0n, 128, lda);
        tdm_load_2d((unsigned)(size_t)&lsB[buf ^ 1][0],
                    Bt + (size_t)blockN * ldb + k0n, 64, ldb);
        __builtin_amdgcn_s_wait_tensorcnt(2);   // pair for step i complete
      } else {
        __builtin_amdgcn_s_wait_tensorcnt(0);
      }
    }
#else
    {
      int k0 = i << 5;
      for (int c2 = 0; c2 < 2; ++c2) {
        int c = tid + c2 * 256;
        int row = c >> 2, col8 = (c & 3) << 3;
        const float4* src = reinterpret_cast<const float4*>(
            &A[(size_t)(blockM + row) * lda + k0 + col8]);
        *reinterpret_cast<float4*>(&lsA[buf][row * 40 + col8]) = *src;
      }
      {
        int row = tid >> 2, col8 = (tid & 3) << 3;
        const float4* src = reinterpret_cast<const float4*>(
            &Bt[(size_t)(blockN + row) * ldb + k0 + col8]);
        *reinterpret_cast<float4*>(&lsB[buf][row * 40 + col8]) = *src;
      }
    }
#endif
    __syncthreads();

    Frag a[2], b[2];
    for (int tm = 0; tm < 2; ++tm) {
      int row = wm * 32 + tm * 16 + (lane & 15);
      for (int v = 0; v < 8; ++v)
        a[tm].u[v] = *reinterpret_cast<const unsigned*>(&lsA[buf][row * 40 + aKmap(v, lane)]);
    }
    for (int tn = 0; tn < 2; ++tn) {
      int row = wn * 32 + tn * 16 + (lane & 15);
      for (int v = 0; v < 8; ++v)
        b[tn].u[v] = *reinterpret_cast<const unsigned*>(&lsB[buf][row * 40 + bKmap(v, lane)]);
    }
    for (int tm = 0; tm < 2; ++tm)
      for (int tn = 0; tn < 2; ++tn)
        acc[tm][tn] = wmma_bf16(a[tm], b[tn], acc[tm][tn]);
    __syncthreads();
  }

  int mhi = (lane >= 16) ? 8 : 0;
  for (int tm = 0; tm < 2; ++tm)
    for (int tn = 0; tn < 2; ++tn)
      for (int r = 0; r < 8; ++r) {
        int gm = blockM + wm * 32 + tm * 16 + r + mhi;      // node
        int gn = blockN + wn * 32 + tn * 16 + (lane & 15);  // bc
        int b = gn / CC, c = gn % CC;
        float val = acc[tm][tn][r];
        if (mode == 0) {
          unsigned short bv = f2bf(val);
          xg[((size_t)gm * BB + b) * KP + kslot + c] = bv;
          outT[(size_t)gn * NN + gm] = bv;
        } else {
          float x0 = bf2f(xcT[(size_t)gn * NN + gm]);
          xg[((size_t)gm * BB + b) * KP + kslot + c] = f2bf(2.f * val - x0);
        }
      }
}

// node-dependent weights: wT[n][o][kp] = sum_d e[n][d] * wp[d][kp][o]  (bf16, zero pad)
__global__ void wgen_kernel(const float* __restrict__ e, const float* __restrict__ wp,
                            unsigned short* __restrict__ wT, int O) {
  size_t idx = (size_t)blockIdx.x * blockDim.x + threadIdx.x;  // (n*O + o)*KP + kp
  int kp = (int)(idx % KP);
  size_t rest = idx / KP;
  int o = (int)(rest % O);
  int n = (int)(rest / O);
  float v = 0.f;
  if (kp < KC) {
    const float* w0 = wp + (size_t)kp * O + o;   // d=0; stride per d = KC*O
    const float* ev = e + n * EE;
    for (int d = 0; d < EE; ++d) v += ev[d] * w0[(size_t)d * KC * O];
  }
  wT[idx] = f2bf(v);
}

__global__ void bias_kernel(const float* __restrict__ e, const float* __restrict__ bp,
                            float* __restrict__ bias, int O) {
  int idx = blockIdx.x * blockDim.x + threadIdx.x;  // n*O + o
  int o = idx % O, n = idx / O;
  float v = 0.f;
  for (int d = 0; d < EE; ++d) v += e[n * EE + d] * bp[d * O + o];
  bias[idx] = v;
}

// per-node GEMM: (64 x KP) @ (KP x O) with WMMA, fused sigmoid / GRU epilogue.
// 8 waves: wave handles mtile = w&3, ntw = (O/16)/2 consecutive n-tiles.
__global__ __launch_bounds__(256) void nodegemm_kernel(const unsigned short* __restrict__ xg,
                                                       const unsigned short* __restrict__ wT,
                                                       const float* __restrict__ bias,
                                                       float* __restrict__ z,
                                                       float* __restrict__ r,
                                                       float* __restrict__ h,
                                                       int O, int phase) {
  int n = blockIdx.x;
  int wave = threadIdx.x >> 5, lane = threadIdx.x & 31;
  int mt = wave & 3;
  int ntw = (O >> 4) >> 1;            // 4 (gate) or 2 (upd)
  int nt0 = (wave >> 2) * ntw;
  const unsigned short* A  = xg + (size_t)n * BB * KP;
  const unsigned short* Bm = wT + (size_t)n * O * KP;

  v8f acc[4];
  for (int j = 0; j < 4; ++j) acc[j] = v8f{};

  int arow = mt * 16 + (lane & 15);
  for (int k0 = 0; k0 < KP; k0 += 32) {
    Frag a;
    for (int v = 0; v < 8; ++v)
      a.u[v] = *reinterpret_cast<const unsigned*>(&A[(size_t)arow * KP + k0 + aKmap(v, lane)]);
    for (int j = 0; j < ntw; ++j) {
      Frag b;
      int brow = (nt0 + j) * 16 + (lane & 15);
      for (int v = 0; v < 8; ++v)
        b.u[v] = *reinterpret_cast<const unsigned*>(&Bm[(size_t)brow * KP + k0 + bKmap(v, lane)]);
      acc[j] = wmma_bf16(a, b, acc[j]);
    }
  }

  int mhi = (lane >= 16) ? 8 : 0;
  for (int j = 0; j < ntw; ++j)
    for (int rr = 0; rr < 8; ++rr) {
      int bidx = mt * 16 + rr + mhi;            // batch index
      int o = (nt0 + j) * 16 + (lane & 15);
      float val = acc[j][rr] + bias[n * O + o];
      if (phase == 0) {
        float s = 1.f / (1.f + __expf(-val));
        size_t zi = ((size_t)bidx * NN + n) * HH;
        if (o < HH) z[zi + o] = s;
        else        r[zi + o - HH] = s;
      } else {
        float hc = tanhf(val);
        size_t hi = ((size_t)bidx * NN + n) * HH + o;
        float rv = r[hi], hold = h[hi];
        h[hi] = rv * hold + (1.f - rv) * hc;
      }
    }
}

// final LN over H + 12x64 conv, output (B, HOR, N, 1)
__global__ void final_kernel(const float* __restrict__ h,
                             const float* __restrict__ lng, const float* __restrict__ lnb,
                             const float* __restrict__ cw, const float* __restrict__ cb,
                             float* __restrict__ out) {
  int idx = blockIdx.x * blockDim.x + threadIdx.x;  // b*NN + n
  if (idx >= BB * NN) return;
  int b = idx >> 10, n = idx & (NN - 1);
  const float* hp = h + (size_t)idx * HH;
  float m = 0.f;
  for (int i = 0; i < HH; ++i) m += hp[i];
  m *= (1.f / HH);
  float v = 0.f;
  for (int i = 0; i < HH; ++i) { float d = hp[i] - m; v += d * d; }
  v *= (1.f / HH);
  float inv = rsqrtf(v + 1e-12f);
  float y[HH];
  for (int i = 0; i < HH; ++i) y[i] = (hp[i] - m) * inv * lng[i] + lnb[i];
  for (int o = 0; o < TT; ++o) {
    float s = cb[o];
    for (int i = 0; i < HH; ++i) s += y[i] * cw[o * HH + i];
    out[((size_t)(b * TT + o)) * NN + n] = s;
  }
}

// ---------------- launcher ----------------
extern "C" void kernel_launch(void* const* d_in, const int* in_sizes, int n_in,
                              void* d_out, int out_size, void* d_ws, size_t ws_size,
                              hipStream_t stream) {
  (void)in_sizes; (void)n_in; (void)out_size; (void)ws_size;
  const float* source   = (const float*)d_in[0];
  const float* node_emb = (const float*)d_in[1];
  const float* time_emb = (const float*)d_in[2];
  const float* gate_wp  = (const float*)d_in[3];
  const float* gate_bp  = (const float*)d_in[4];
  const float* gate_lng = (const float*)d_in[5];
  const float* gate_lnb = (const float*)d_in[6];
  const float* upd_wp   = (const float*)d_in[7];
  const float* upd_bp   = (const float*)d_in[8];
  const float* upd_lng  = (const float*)d_in[9];
  const float* upd_lnb  = (const float*)d_in[10];
  const float* out_lng  = (const float*)d_in[11];
  const float* out_lnb  = (const float*)d_in[12];
  const float* conv_w   = (const float*)d_in[13];
  const float* conv_b   = (const float*)d_in[14];
  float* out = (float*)d_out;

  char* ws = (char*)d_ws;
  size_t off = 0;
  auto alloc = [&](size_t bytes) -> void* {
    void* p = ws + off;
    off = (off + bytes + 255) & ~(size_t)255;
    return p;
  };
  float*          h      = (float*)alloc(sizeof(float) * BB * NN * HH);
  float*          z      = (float*)alloc(sizeof(float) * BB * NN * HH);
  float*          r      = (float*)alloc(sizeof(float) * BB * NN * HH);
  float*          eg     = (float*)alloc(sizeof(float) * NN * EE);
  float*          eu     = (float*)alloc(sizeof(float) * NN * EE);
  unsigned short* egp    = (unsigned short*)alloc(2ull * NN * 32);
  unsigned short* eup    = (unsigned short*)alloc(2ull * NN * 32);
  float*          logits = (float*)alloc(sizeof(float) * NN * NN);
  unsigned short* sup    = (unsigned short*)alloc(2ull * NN * NN);
  unsigned short* xcT    = (unsigned short*)alloc(2ull * BC * NN);
  unsigned short* xg1T   = (unsigned short*)alloc(2ull * BC * NN);
  unsigned short* xg     = (unsigned short*)alloc(2ull * NN * BB * KP);
  unsigned short* wT     = (unsigned short*)alloc(2ull * NN * 128 * KP);
  float*          bias   = (float*)alloc(sizeof(float) * NN * 128);

  zero_f<<<(BB * NN * HH + 255) / 256, 256, 0, stream>>>(h, BB * NN * HH);

  for (int t = 0; t < TT; ++t) {
    e_kernel<<<NN / 256, 256, 0, stream>>>(node_emb, time_emb + t * EE,
                                           gate_lng, gate_lnb, upd_lng, upd_lnb,
                                           eg, eu, egp, eup);
    for (int phase = 0; phase < 2; ++phase) {
      const unsigned short* ep = phase ? eup : egp;
      const float* ef = phase ? eu : eg;
      const float* wp = phase ? upd_wp : gate_wp;
      const float* bp = phase ? upd_bp : gate_bp;
      const int O = phase ? HH : 2 * HH;

      logits_kernel<<<512, 256, 0, stream>>>(ep, logits);
      softmax_kernel<<<NN, 256, 0, stream>>>(logits, sup);
      pack_kernel<<<(BC * NN) / 256, 256, 0, stream>>>(source, t, h, z, phase, xcT, xg);

      dim3 g1(BC / 64, NN / 128);
      // xg1 = sup @ X  -> xg slot k=1 (Kidx base CC) and transposed copy
      gemm_nt<<<g1, 256, 0, stream>>>(sup, NN, xcT, NN, NN, 0, nullptr, xg, xg1T, CC);
      // xg2 = 2*sup@xg1 - X -> xg slot k=2 (Kidx base 2*CC)
      gemm_nt<<<g1, 256, 0, stream>>>(sup, NN, xg1T, NN, NN, 1, xcT, xg, nullptr, 2 * CC);

      wgen_kernel<<<(int)(((size_t)NN * O * KP) / 256), 256, 0, stream>>>(ef, wp, wT, O);
      bias_kernel<<<(NN * O) / 256, 256, 0, stream>>>(ef, bp, bias, O);
      nodegemm_kernel<<<NN, 256, 0, stream>>>(xg, wT, bias, z, r, h, O, phase);
    }
  }

  final_kernel<<<(BB * NN) / 256, 256, 0, stream>>>(h, out_lng, out_lnb, conv_w, conv_b, out);
}
